// HSRnnForCausalLM_6141803233965
// MI455X (gfx1250) — matrix operand order repaired and checked by reading, not compiled, not run on hardware
//
#include <hip/hip_runtime.h>
#include <hip/hip_bf16.h>

// ---------------------------------------------------------------------------
// DualStateLIF 3-layer scan + vocab-head GEMM for MI455X (gfx1250, wave32)
//   Phase 1: persistent 1-block scan, WMMA bf16, x-staging overlapped w/ L2,
//            weights register-resident except layer-2 kc{6,7} (LDS) to stay
//            under the 256-VGPR ceiling (no scratch spills in the hot loop)
//   Phase 2: head GEMM [8192x256]x[256x32000], TDM-staged A tile in LDS
// ---------------------------------------------------------------------------

typedef __attribute__((ext_vector_type(16))) __bf16          v16bf;
typedef __attribute__((ext_vector_type(8)))  float           v8f;
typedef __attribute__((ext_vector_type(16))) unsigned short  v16u;
typedef __attribute__((ext_vector_type(8)))  unsigned short  v8u;
typedef __attribute__((ext_vector_type(4)))  unsigned int    v4ui;
typedef __attribute__((ext_vector_type(8)))  int             v8i;
typedef __attribute__((ext_vector_type(4)))  int             v4i;

#define LIF_ALPHA  0.9f
#define LIF_BETA   0.95f
#define LIF_THRESH 1.0f
#define LIF_SLOPE  4.0f

#define NB   4      // batch
#define NT   2048   // timesteps
#define NH   256    // hidden (== D)
#define NV   32000  // vocab

// round-to-nearest-even f32 -> bf16 bits
static __device__ __forceinline__ unsigned short f2bf(float f) {
  unsigned u = __builtin_bit_cast(unsigned, f);
  unsigned r = u + 0x7FFFu + ((u >> 16) & 1u);
  return (unsigned short)(r >> 16);
}

static __device__ __forceinline__ v16bf as_bf(v16u x) {
  union { v16u u; v16bf b; } c;
  c.u = x;
  return c.b;
}

union AFrag {
  struct { v8u lo; v8u hi; } p;  // halfs [0..7] and [8..15]
  v16u v;
};

// ---------------------------------------------------------------------------
// Phase 0: f32 -> bf16 weight conversion (W_head)
// ---------------------------------------------------------------------------
__global__ void cvt_bf16_kernel(const float* __restrict__ in,
                                unsigned short* __restrict__ out, int n) {
  int i = blockIdx.x * blockDim.x + threadIdx.x;
  if (i < n) out[i] = f2bf(in[i]);
}

// ---------------------------------------------------------------------------
// Phase 1: sequential LIF scan. One block, 16 waves; wave w owns hidden
// columns [16w, 16w+16). Per layer: 8x v_wmma_f32_16x16x32_bf16.
// Weight residency: L0/L1 and L2 kc<6 in VGPRs; L2 kc{6,7} in LDS (loaded
// early in the layer-2 chain, consumed last). LIF state (i,v) in VGPRs of
// lanes 0..15. Buffer parity: x_t in buf[t&1]; L0: A->B, L1: B->A, L2 reads
// A and writes only global, so x_{t+1} staging overlaps layer-2 compute.
// ---------------------------------------------------------------------------
__global__ __launch_bounds__(512) void lif_scan_kernel(
    const float* __restrict__ x,                                  // [B,T,256]
    const float* __restrict__ W0, const float* __restrict__ b0,
    const float* __restrict__ W1, const float* __restrict__ b1,
    const float* __restrict__ W2, const float* __restrict__ b2,
    unsigned short* __restrict__ hs)                              // [B*T,256] bf16
{
  __shared__ __align__(32) unsigned short hbuf[2][16][NH];      // 16 KB ping-pong
  __shared__ __align__(32) unsigned short wlds[16][2][32][16];  // 32 KB: L2 kc{6,7}

  const int tid   = threadIdx.x;
  const int lane  = tid & 31;
  const int wave  = tid >> 5;
  const int n0    = wave << 4;          // this wave's N-tile base
  const int lm    = lane & 15;
  const int lh    = (lane < 16) ? 0 : 1;
  const int col   = n0 + lm;            // hidden unit owned by this lane (lanes<16)

  // zero both LDS buffers once (rows 4..15 must stay zero: M padding)
  for (int i = tid; i < 2 * 16 * NH; i += 512)
    ((unsigned short*)hbuf)[i] = 0;

  // --- preload weight fragments (bf16) ------------------------------------
  // B-matrix 32x16 layout: lane = column n, 16 contiguous K per lane,
  // lanes 0-15 hold K [32kc, 32kc+16), lanes 16-31 hold K [32kc+16, 32kc+32).
  v16u wB01[2][8];  // layers 0,1: all K-chunks in registers
  v16u wB2[6];      // layer 2: kc 0..5 in registers, kc 6,7 -> LDS
  {
    const float* Ws[3] = {W0, W1, W2};
#pragma unroll
    for (int l = 0; l < 3; ++l) {
      const float* W = Ws[l];
#pragma unroll
      for (int kc = 0; kc < 8; ++kc) {
        const float* src = W + col * NH + kc * 32 + lh * 16;
        v16u r;
#pragma unroll
        for (int i = 0; i < 16; ++i) r[i] = f2bf(src[i]);
        if (l < 2)            wB01[l][kc] = r;
        else if (kc < 6)      wB2[kc] = r;
        else                  *(v16u*)&wlds[wave][kc - 6][lane][0] = r;
      }
    }
  }

  const float bias[3] = {b0[col], b1[col], b2[col]};
  float si[3][4] = {};   // synaptic current per (layer, batch)
  float sv[3][4] = {};   // membrane voltage per (layer, batch)

  // x-staging indices: 1024 f32 over 512 threads (2 each)
  const int xb0 = tid >> 8,          xd0 = tid & (NH - 1);
  const int xb1 = (tid + 512) >> 8,  xd1 = (tid + 512) & (NH - 1);

  // stage x_0 into buffer 0
  hbuf[0][xb0][xd0] = f2bf(x[((size_t)xb0 * NT) * NH + xd0]);
  hbuf[0][xb1][xd1] = f2bf(x[((size_t)xb1 * NT) * NH + xd1]);
  __syncthreads();

  for (int t = 0; t < NT; ++t) {
    const int pb = t & 1;
#pragma unroll
    for (int l = 0; l < 3; ++l) {
      const int rd = (l == 1) ? (pb ^ 1) : pb;   // L0:pb, L1:pb^1, L2:pb
      const int wr = (l == 0) ? (pb ^ 1) : pb;   // L2 writes no LDS

      // issue next-timestep x loads early, hidden under layer-2 WMMA
      float xv0 = 0.0f, xv1 = 0.0f;
      const bool stage = (l == 2) && (t + 1 < NT);
      if (stage) {
        xv0 = x[((size_t)xb0 * NT + (t + 1)) * NH + xd0];
        xv1 = x[((size_t)xb1 * NT + (t + 1)) * NH + xd1];
      }

      // layer-2's LDS-resident tail fragments: load first, consume last
      v16u ldsB0, ldsB1;
      if (l == 2) {
        ldsB0 = *(const v16u*)&wlds[wave][0][lane][0];
        ldsB1 = *(const v16u*)&wlds[wave][1][lane][0];
      }

      // ---- GEMM tile: acc[16x16] = h[16x256] @ W_l^T[:, n0:n0+16] --------
      v8f acc = {};
      const unsigned short* row = &hbuf[rd][lm][0];
#pragma unroll
      for (int kc = 0; kc < 8; ++kc) {
        // A 16x32 layout: lanes<16 hold K [32kc, 32kc+16),
        //                 lanes>=16 hold K [32kc+16, 32kc+32)
        AFrag a;
        a.p.lo = *(const v8u*)(row + kc * 32 + lh * 16);
        a.p.hi = *(const v8u*)(row + kc * 32 + lh * 16 + 8);
        v16u bfrag;
        if (l < 2)        bfrag = wB01[l][kc];
        else if (kc < 6)  bfrag = wB2[kc];
        else              bfrag = (kc == 6) ? ldsB0 : ldsB1;
        acc = __builtin_amdgcn_wmma_f32_16x16x32_bf16(
            false, as_bf(a.v), false, as_bf(bfrag), (short)0, acc,
            false, false);
      }
      // ---- LIF state update (lanes 0..15 hold rows 0..3 in acc[0..3]) ----
      if (lane < 16) {
#pragma unroll
        for (int b = 0; b < 4; ++b) {
          float g = acc[b] + bias[l];
          si[l][b] = LIF_ALPHA * si[l][b] + g;
          sv[l][b] = LIF_BETA * sv[l][b] + si[l][b];
          float e = __expf(-LIF_SLOPE * (sv[l][b] - LIF_THRESH));
          float sp = __builtin_amdgcn_rcpf(1.0f + e);   // fast v_rcp_f32
          sv[l][b] -= sp * LIF_THRESH;  // soft reset
          unsigned short sb = f2bf(sp);
          if (l < 2) hbuf[wr][b][col] = sb;          // feed next layer
          else hs[((size_t)b * NT + t) * NH + col] = sb;  // head row b*T+t
        }
      }
      if (stage) {  // x_{t+1} -> buf pb^1 (dead after L1's reads)
        hbuf[pb ^ 1][xb0][xd0] = f2bf(xv0);
        hbuf[pb ^ 1][xb1][xd1] = f2bf(xv1);
      }
      __syncthreads();
    }
  }
}

// ---------------------------------------------------------------------------
// Phase 2: head GEMM. out[m, n] = hs[m, :] @ W_head[n, :] + b_head[n]
// Block = 4 waves covering 16(M) x 256(N); the shared 16x256 bf16 A tile is
// DMA'd into LDS once per block by the Tensor Data Mover (async, TENSORcnt),
// removing 4x-redundant L2 reads. B (16MB) stays L2-resident. Kernel is
// bounded by the 1 GB f32 logit store (~45us at 23.3 TB/s).
// ---------------------------------------------------------------------------
__global__ __launch_bounds__(128) void head_gemm_kernel(
    const unsigned short* __restrict__ hs,   // [8192,256] bf16
    const unsigned short* __restrict__ Wb,   // [32000,256] bf16
    const float* __restrict__ bh,            // [32000]
    float* __restrict__ out)                 // [8192,32000] f32
{
  __shared__ __align__(32) unsigned short sA[16][NH];  // 8 KB, LDS offset 0

  const int lane = threadIdx.x & 31;
  const int wave = threadIdx.x >> 5;
  const int lm   = lane & 15;
  const int lh   = (lane < 16) ? 0 : 1;
  const int m0   = blockIdx.y << 4;
  const int n0   = blockIdx.x * 256 + wave * 64;

#if __has_builtin(__builtin_amdgcn_tensor_load_to_lds)
  // --- TDM: async DMA of the 16x256 bf16 tile (tile row-major, 2B elems) --
  if (wave == 0) {
    const unsigned long long ga = (unsigned long long)(hs + (size_t)m0 * NH);
    // D# group 0: count=1, lds_addr=0, global_addr, type=2 ("image")
    v4ui g0 = {1u, 0u, (unsigned)ga,
               ((unsigned)(ga >> 32) & 0x01FFFFFFu) | 0x80000000u};
    // D# group 1: data_size=2B(code 1), tensor_dim0=256, tensor_dim1=8192,
    //             tile_dim0=256, tile_dim1=16, tensor_dim0_stride=256
    v8i g1 = {0x00010000, 0x01000000, 0x20000000, 0x01000000, 16, 256, 0, 0};
    v4i gz = {0, 0, 0, 0};
#if __clang_major__ >= 23
    v8i gz8 = {0, 0, 0, 0, 0, 0, 0, 0};
    __builtin_amdgcn_tensor_load_to_lds(g0, g1, gz, gz, gz8, 0);
#else
    __builtin_amdgcn_tensor_load_to_lds(g0, g1, gz, gz, 0);
#endif
    __builtin_amdgcn_s_wait_tensorcnt(0);
  }
#else
  // fallback: cooperative vector copy (4x b128 per thread)
  {
    const v8u* src = (const v8u*)(hs + (size_t)m0 * NH);
    v8u* dst = (v8u*)sA;
    for (int i = threadIdx.x; i < (16 * NH) / 8; i += 128) dst[i] = src[i];
  }
#endif
  __syncthreads();

  v8f acc[4] = {{}, {}, {}, {}};
  const unsigned short* arow = &sA[lm][0];

#pragma unroll
  for (int kc = 0; kc < 8; ++kc) {
    AFrag a;
    a.p.lo = *(const v8u*)(arow + kc * 32 + lh * 16);
    a.p.hi = *(const v8u*)(arow + kc * 32 + lh * 16 + 8);
    v16bf av = as_bf(a.v);
#pragma unroll
    for (int j = 0; j < 4; ++j) {
      const unsigned short* bp =
          Wb + (size_t)(n0 + j * 16 + lm) * NH + kc * 32 + lh * 16;
      acc[j] = __builtin_amdgcn_wmma_f32_16x16x32_bf16(
          false, av, false, as_bf(*(const v16u*)bp), (short)0, acc[j],
          false, false);
    }
  }

  const long mbase = m0 + ((lane < 16) ? 0 : 8);
#pragma unroll
  for (int j = 0; j < 4; ++j) {
    const int n = n0 + j * 16 + lm;
    const float bias = bh[n];
    float* o = out + n;
#pragma unroll
    for (int r = 0; r < 8; ++r)
      o[(mbase + r) * (long)NV] = acc[j][r] + bias;
  }
}

// ---------------------------------------------------------------------------
extern "C" void kernel_launch(void* const* d_in, const int* in_sizes, int n_in,
                              void* d_out, int out_size, void* d_ws,
                              size_t ws_size, hipStream_t stream) {
  (void)in_sizes; (void)n_in; (void)out_size; (void)ws_size;

  const float* x      = (const float*)d_in[0];
  const float* W0     = (const float*)d_in[1];
  const float* b0     = (const float*)d_in[2];
  const float* W1     = (const float*)d_in[3];
  const float* b1     = (const float*)d_in[4];
  const float* W2     = (const float*)d_in[5];
  const float* b2     = (const float*)d_in[6];
  const float* W_head = (const float*)d_in[7];
  const float* b_head = (const float*)d_in[8];

  // workspace layout: [0,4MB) hs bf16 [8192x256]; [4MB, ~20.6MB) W_head bf16
  unsigned short* hs = (unsigned short*)d_ws;
  unsigned short* Wb = (unsigned short*)((char*)d_ws + (size_t)4 * 1024 * 1024);

  // convert head weights to bf16
  {
    const int n = NV * NH;
    cvt_bf16_kernel<<<(n + 255) / 256, 256, 0, stream>>>(W_head, Wb, n);
  }

  // sequential LIF scan: one persistent block, 16 waves
  lif_scan_kernel<<<1, 512, 0, stream>>>(x, W0, b0, W1, b1, W2, b2, hs);

  // head GEMM: grid (32000/256, 8192/16)
  head_gemm_kernel<<<dim3(NV / 256, (NB * NT) / 16), 128, 0, stream>>>(
      hs, Wb, b_head, (float*)d_out);
}
